// GATDrugEncoder_71734543777907
// MI455X (gfx1250) — matrix operand order repaired and checked by reading, not compiled, not run on hardware
//
#include <hip/hip_runtime.h>
#include <hip/hip_bf16.h>

// ---------------- problem constants (from reference) ----------------
#define N_NODES 100000
#define N_EDGES 1600000
#define B_GRAPHS 512
#define HID 128
#define EPS_BN 1e-5f

typedef _Float16 v16h_t __attribute__((ext_vector_type(16)));
typedef float    v8f_t  __attribute__((ext_vector_type(8)));

// order-preserving float atomic max (works for mixed signs)
__device__ __forceinline__ void atomic_max_f32(float* addr, float val) {
  if (!(val < 0.0f)) {
    atomicMax((int*)addr, __float_as_int(val));
  } else {
    atomicMin((unsigned int*)addr, (unsigned int)__float_as_int(val));
  }
}

// ---------------- small utility kernels ----------------
__global__ void fill_f32(float* __restrict__ p, float v, int n) {
  int i = blockIdx.x * blockDim.x + threadIdx.x;
  if (i < n) p[i] = v;
}

__global__ void cvt_f32_to_f16(const float* __restrict__ s, _Float16* __restrict__ d, int n) {
  int i = blockIdx.x * blockDim.x + threadIdx.x;
  if (i < n) d[i] = (_Float16)s[i];
}

// ---------------- WMMA GEMM: C[M,NC] = act(A[M,K]_f32 * B[K,NC]_f16 + bias) ----
// grid.x = M/16 (M multiple of 16), blockDim.x = (NC/16)*32; one 16x16 tile/wave
// act: 0 = none, 1 = tanh
__global__ void wmma_gemm_f16(const float* __restrict__ A,
                              const _Float16* __restrict__ Bh,
                              const float* __restrict__ bias,
                              float* __restrict__ C,
                              int K, int NC, int act) {
  const int wave  = threadIdx.x >> 5;
  const int lane  = threadIdx.x & 31;
  const int tileM = blockIdx.x * 16;
  const int tileN = wave * 16;

  v8f_t acc = {};
  const int rowA = tileM + (lane & 15);
  const int koff = (lane >> 4) << 3;                 // 0 or 8 per ISA A-layout
  const float* aBase = A + (size_t)rowA * K + koff;

  for (int k0 = 0; k0 < K; k0 += 32) {
    // A fragment: 16-bit A 16x32 layout (lane<16: K {0..7,16..23}; lane>=16: +8)
    v16h_t afrag;
    const float* ap = aBase + k0;
#pragma unroll
    for (int i = 0; i < 8; ++i) afrag[i] = (_Float16)ap[i];
#pragma unroll
    for (int i = 0; i < 8; ++i) afrag[8 + i] = (_Float16)ap[16 + i];

    // B fragment: lane holds column tileN+lane%16, K {0..7,16..23} (+8 for lane>=16)
    v16h_t bfrag;
    const _Float16* bp = Bh + (size_t)(k0 + koff) * NC + tileN + (lane & 15);
#pragma unroll
    for (int i = 0; i < 8; ++i) bfrag[i] = bp[(size_t)i * NC];
#pragma unroll
    for (int i = 0; i < 8; ++i) bfrag[8 + i] = bp[(size_t)(16 + i) * NC];

    acc = __builtin_amdgcn_wmma_f32_16x16x32_f16(false, afrag, false, bfrag,
                                                 (short)0, acc, false, false);
  }

  // D layout: VGPR r, lane L -> row r + 8*(L>=16), col L%16
  const int col   = tileN + (lane & 15);
  const int rbase = tileM + ((lane >> 4) << 3);
  const float bv = bias ? bias[col] : 0.0f;
#pragma unroll
  for (int r = 0; r < 8; ++r) {
    float v = acc[r] + bv;
    if (act == 1) v = tanhf(v);
    C[(size_t)(rbase + r) * NC + col] = v;
  }
}

// ---------------- GAT attention kernels ----------------
// al_s[n*H+h] = dot(xs[n, h*CH : h*CH+CH], a_src[h]);  same for al_d
__global__ void node_scores(const float* __restrict__ xs,
                            const float* __restrict__ a_src,
                            const float* __restrict__ a_dst,
                            float* __restrict__ al_s, float* __restrict__ al_d,
                            int H, int CH) {
  int idx = blockIdx.x * blockDim.x + threadIdx.x;
  if (idx >= N_NODES * H) return;
  int n = idx / H, h = idx - n * H;
  const float* xp = xs + (size_t)n * HID + h * CH;
  float ss = 0.f, sd = 0.f;
  for (int c = 0; c < CH; ++c) {
    float v = xp[c];
    ss += v * a_src[h * CH + c];
    sd += v * a_dst[h * CH + c];
  }
  al_s[idx] = ss;
  al_d[idx] = sd;
}

__device__ __forceinline__ void edge_endpoints(int e, const int* SRC, const int* DST,
                                               int& s, int& d) {
  if (e < N_EDGES) { s = SRC[e]; d = DST[e]; }
  else             { s = e - N_EDGES; d = s; }   // self loops appended
}

__global__ void edge_max(const int* __restrict__ SRC, const int* __restrict__ DST,
                         const float* __restrict__ al_s, const float* __restrict__ al_d,
                         float* __restrict__ m, int H) {
  int idx = blockIdx.x * blockDim.x + threadIdx.x;
  int total = (N_EDGES + N_NODES) * H;
  if (idx >= total) return;
  int e = idx / H, h = idx - e * H;
  int s, d; edge_endpoints(e, SRC, DST, s, d);
  float v = al_s[s * H + h] + al_d[d * H + h];
  v = v > 0.f ? v : 0.2f * v;                    // leaky relu 0.2
  atomic_max_f32(&m[d * H + h], v);
}

// ee = exp(e - m[dst]); store per (edge,head) and accumulate z[dst]
__global__ void edge_expsum(const int* __restrict__ SRC, const int* __restrict__ DST,
                            const float* __restrict__ al_s, const float* __restrict__ al_d,
                            const float* __restrict__ m, float* __restrict__ z,
                            float* __restrict__ ab, int H) {
  int idx = blockIdx.x * blockDim.x + threadIdx.x;
  int total = (N_EDGES + N_NODES) * H;
  if (idx >= total) return;
  int e = idx / H, h = idx - e * H;
  int s, d; edge_endpoints(e, SRC, DST, s, d);
  float v = al_s[s * H + h] + al_d[d * H + h];
  v = v > 0.f ? v : 0.2f * v;
  float ee = __expf(v - m[d * H + h]);
  ab[idx] = ee;
  atomicAdd(&z[d * H + h], ee);
}

// alpha = ee / z[dst]
__global__ void edge_alpha(const int* __restrict__ DST, const float* __restrict__ z,
                           float* __restrict__ ab, int H) {
  int idx = blockIdx.x * blockDim.x + threadIdx.x;
  int total = (N_EDGES + N_NODES) * H;
  if (idx >= total) return;
  int e = idx / H, h = idx - e * H;
  int d = (e < N_EDGES) ? DST[e] : (e - N_EDGES);
  ab[idx] = ab[idx] / (z[d * H + h] + 1e-16f);
}

// agg[dst, c..c+3] += alpha * xs[src, c..c+3]; one thread per (edge, 4 channels)
__global__ void edge_agg(const int* __restrict__ SRC, const int* __restrict__ DST,
                         const float* __restrict__ ab,
                         const float* __restrict__ xs, float* __restrict__ agg,
                         int H, int log2ch) {
  int idx = blockIdx.x * blockDim.x + threadIdx.x;
  int total = (N_EDGES + N_NODES) * (HID / 4);
  if (idx >= total) return;
  int c4 = idx & 31;           // HID/4 = 32 groups
  int e  = idx >> 5;
  int c  = c4 << 2;
  int h  = c >> log2ch;        // CH=32 -> 5, CH=128 -> 7 (4 channels never straddle a head)
  int s, d; edge_endpoints(e, SRC, DST, s, d);
  float alpha = ab[e * H + h];
  float4 xv = reinterpret_cast<const float4*>(xs + (size_t)s * HID)[c4];
  float* ap = agg + (size_t)d * HID + c;
  atomicAdd(ap + 0, alpha * xv.x);
  atomicAdd(ap + 1, alpha * xv.y);
  atomicAdd(ap + 2, alpha * xv.z);
  atomicAdd(ap + 3, alpha * xv.w);
}

// h += elu(bn(agg + conv_bias))  -- float4 per thread
__global__ void bn_elu_res(float* __restrict__ h, const float* __restrict__ agg,
                           const float* __restrict__ cbias,
                           const float* __restrict__ g, const float* __restrict__ b,
                           const float* __restrict__ rm, const float* __restrict__ rv) {
  int idx = blockIdx.x * blockDim.x + threadIdx.x;
  if (idx >= N_NODES * (HID / 4)) return;
  int c = (idx << 2) & (HID - 1);
  float4 a  = reinterpret_cast<const float4*>(agg)[idx];
  float4 hv = reinterpret_cast<float4*>(h)[idx];
  float av[4] = {a.x, a.y, a.z, a.w};
  float hvv[4] = {hv.x, hv.y, hv.z, hv.w};
#pragma unroll
  for (int j = 0; j < 4; ++j) {
    int cc = c + j;
    float v = av[j] + cbias[cc];
    float scale = g[cc] * rsqrtf(rv[cc] + EPS_BN);
    v = (v - rm[cc]) * scale + b[cc];
    v = v > 0.f ? v : (__expf(v) - 1.0f);
    hvv[j] += v;
  }
  reinterpret_cast<float4*>(h)[idx] = make_float4(hvv[0], hvv[1], hvv[2], hvv[3]);
}

// ---------------- readout kernels ----------------
__global__ void ro_scores(const float* __restrict__ t, const float* __restrict__ w2,
                          const float* __restrict__ b2, float* __restrict__ scores) {
  int n = blockIdx.x * blockDim.x + threadIdx.x;
  if (n >= N_NODES) return;
  const float* tp = t + (size_t)n * 64;
  float s = b2[0];
  for (int i = 0; i < 64; ++i) s += tp[i] * w2[i];
  scores[n] = s;
}

__global__ void seg_max(const float* __restrict__ scores, const int* __restrict__ batch,
                        float* __restrict__ gm) {
  int n = blockIdx.x * blockDim.x + threadIdx.x;
  if (n >= N_NODES) return;
  atomic_max_f32(&gm[batch[n]], scores[n]);
}

__global__ void seg_exp(const float* __restrict__ scores, const int* __restrict__ batch,
                        const float* __restrict__ gm, float* __restrict__ gz,
                        float* __restrict__ attn) {
  int n = blockIdx.x * blockDim.x + threadIdx.x;
  if (n >= N_NODES) return;
  int bidx = batch[n];
  float e = __expf(scores[n] - gm[bidx]);
  attn[n] = e;
  atomicAdd(&gz[bidx], e);
}

__global__ void seg_norm(float* __restrict__ attn, const int* __restrict__ batch,
                         const float* __restrict__ gz) {
  int n = blockIdx.x * blockDim.x + threadIdx.x;
  if (n >= N_NODES) return;
  attn[n] = attn[n] / gz[batch[n]];
}

// emb[batch[n], c..c+3] += attn[n] * h[n, c..c+3]
__global__ void pool(const float* __restrict__ h, const float* __restrict__ attn,
                     const int* __restrict__ batch, float* __restrict__ emb) {
  int idx = blockIdx.x * blockDim.x + threadIdx.x;
  if (idx >= N_NODES * (HID / 4)) return;
  int c4 = idx & 31;
  int n  = idx >> 5;
  float a = attn[n];
  int bidx = batch[n];
  float4 hv = reinterpret_cast<const float4*>(h + (size_t)n * HID)[c4];
  float* ep = emb + (size_t)bidx * HID + (c4 << 2);
  atomicAdd(ep + 0, a * hv.x);
  atomicAdd(ep + 1, a * hv.y);
  atomicAdd(ep + 2, a * hv.z);
  atomicAdd(ep + 3, a * hv.w);
}

// ---------------- driver ----------------
static inline int cdiv(long long a, long long b) { return (int)((a + b - 1) / b); }

extern "C" void kernel_launch(void* const* d_in, const int* in_sizes, int n_in,
                              void* d_out, int out_size, void* d_ws, size_t ws_size,
                              hipStream_t stream) {
  (void)in_sizes; (void)n_in; (void)out_size; (void)ws_size;

  const float* x            = (const float*)d_in[0];
  const int*   ei           = (const int*)  d_in[1];
  const int*   SRC          = ei;
  const int*   DST          = ei + N_EDGES;
  const int*   batch        = (const int*)  d_in[2];
  const float* W_in         = (const float*)d_in[3];
  const float* b_in         = (const float*)d_in[4];
  const float* lin_w        = (const float*)d_in[5];
  const float* att_src      = (const float*)d_in[6];
  const float* att_dst      = (const float*)d_in[7];
  const float* conv_b       = (const float*)d_in[8];
  const float* lin_w_last   = (const float*)d_in[9];
  const float* att_src_last = (const float*)d_in[10];
  const float* att_dst_last = (const float*)d_in[11];
  const float* b_last       = (const float*)d_in[12];
  const float* bn_g         = (const float*)d_in[13];
  const float* bn_b         = (const float*)d_in[14];
  const float* bn_m         = (const float*)d_in[15];
  const float* bn_v         = (const float*)d_in[16];
  const float* ro_w1        = (const float*)d_in[17];
  const float* ro_b1        = (const float*)d_in[18];
  const float* ro_w2        = (const float*)d_in[19];
  const float* ro_b2        = (const float*)d_in[20];

  float* out  = (float*)d_out;
  float* emb  = out;                           // [512,128]
  float* attn = out + (size_t)B_GRAPHS * HID;  // [100000]

  // ---- workspace layout ----
  const size_t NH = (size_t)N_NODES * HID;
  float* h    = (float*)d_ws;
  float* xs   = h   + NH;                      // reused as tanh intermediate [N,64]
  float* agg  = xs  + NH;
  float* al_s = agg + NH;                      // [N, 4] max
  float* al_d = al_s + (size_t)N_NODES * 4;
  float* mbuf = al_d + (size_t)N_NODES * 4;
  float* zbuf = mbuf + (size_t)N_NODES * 4;
  float* abuf = zbuf + (size_t)N_NODES * 4;    // alpha per (edge,head), max H=4
  float* scr  = abuf + (size_t)(N_EDGES + N_NODES) * 4;  // scores [N]
  float* gm   = scr  + N_NODES;                // [B]
  float* gz   = gm   + B_GRAPHS;               // [B]
  _Float16* wh = (_Float16*)(gz + B_GRAPHS);
  _Float16* wh_Win  = wh;                      // 128*128
  _Float16* wh_lin  = wh_Win + 128 * 128;      // 2*128*128
  _Float16* wh_last = wh_lin + 2 * 128 * 128;  // 128*128
  _Float16* wh_ro1  = wh_last + 128 * 128;     // 128*64

  const int TB = 256;
  const long long ENH4 = (long long)(N_EDGES + N_NODES) * 4;
  const long long EN32 = (long long)(N_EDGES + N_NODES) * (HID / 4);

  // ---- weight conversion to f16 ----
  cvt_f32_to_f16<<<cdiv(128 * 128, TB), TB, 0, stream>>>(W_in, wh_Win, 128 * 128);
  cvt_f32_to_f16<<<cdiv(2 * 128 * 128, TB), TB, 0, stream>>>(lin_w, wh_lin, 2 * 128 * 128);
  cvt_f32_to_f16<<<cdiv(128 * 128, TB), TB, 0, stream>>>(lin_w_last, wh_last, 128 * 128);
  cvt_f32_to_f16<<<cdiv(128 * 64, TB), TB, 0, stream>>>(ro_w1, wh_ro1, 128 * 64);

  // ---- input projection: h = x @ W_in + b_in ----
  wmma_gemm_f16<<<N_NODES / 16, 256, 0, stream>>>(x, wh_Win, b_in, h, 128, 128, 0);

  // ---- three GAT layers ----
  for (int i = 0; i < 3; ++i) {
    const _Float16* W  = (i < 2) ? (wh_lin + (size_t)i * 128 * 128) : wh_last;
    const float*   a_s = (i < 2) ? (att_src + (size_t)i * 128) : att_src_last;
    const float*   a_d = (i < 2) ? (att_dst + (size_t)i * 128) : att_dst_last;
    const float*   cb  = (i < 2) ? (conv_b + (size_t)i * 128) : b_last;
    const int H      = (i < 2) ? 4 : 1;
    const int CH     = HID / H;
    const int log2ch = (i < 2) ? 5 : 7;
    const long long ENH = (long long)(N_EDGES + N_NODES) * H;

    wmma_gemm_f16<<<N_NODES / 16, 256, 0, stream>>>(h, W, nullptr, xs, 128, 128, 0);
    node_scores<<<cdiv((long long)N_NODES * H, TB), TB, 0, stream>>>(xs, a_s, a_d, al_s, al_d, H, CH);
    fill_f32<<<cdiv((long long)N_NODES * H, TB), TB, 0, stream>>>(mbuf, -3e38f, N_NODES * H);
    fill_f32<<<cdiv((long long)N_NODES * H, TB), TB, 0, stream>>>(zbuf, 0.0f, N_NODES * H);
    edge_max<<<cdiv(ENH, TB), TB, 0, stream>>>(SRC, DST, al_s, al_d, mbuf, H);
    edge_expsum<<<cdiv(ENH, TB), TB, 0, stream>>>(SRC, DST, al_s, al_d, mbuf, zbuf, abuf, H);
    edge_alpha<<<cdiv(ENH, TB), TB, 0, stream>>>(DST, zbuf, abuf, H);
    fill_f32<<<cdiv((long long)NH, TB), TB, 0, stream>>>(agg, 0.0f, (int)NH);
    edge_agg<<<cdiv(EN32, TB), TB, 0, stream>>>(SRC, DST, abuf, xs, agg, H, log2ch);
    bn_elu_res<<<cdiv((long long)NH / 4, TB), TB, 0, stream>>>(h, agg, cb,
        bn_g + (size_t)i * 128, bn_b + (size_t)i * 128, bn_m + (size_t)i * 128, bn_v + (size_t)i * 128);
  }
  (void)ENH4;

  // ---- attention readout ----
  // t = tanh(h @ ro_w1 + ro_b1)   [N,64]  (reuse xs buffer)
  wmma_gemm_f16<<<N_NODES / 16, 128, 0, stream>>>(h, wh_ro1, ro_b1, xs, 128, 64, 1);
  ro_scores<<<cdiv(N_NODES, TB), TB, 0, stream>>>(xs, ro_w2, ro_b2, scr);

  fill_f32<<<cdiv(B_GRAPHS, TB), TB, 0, stream>>>(gm, -3e38f, B_GRAPHS);
  fill_f32<<<cdiv(B_GRAPHS, TB), TB, 0, stream>>>(gz, 0.0f, B_GRAPHS);
  seg_max<<<cdiv(N_NODES, TB), TB, 0, stream>>>(scr, batch, gm);
  seg_exp<<<cdiv(N_NODES, TB), TB, 0, stream>>>(scr, batch, gm, gz, attn);
  seg_norm<<<cdiv(N_NODES, TB), TB, 0, stream>>>(attn, batch, gz);

  fill_f32<<<cdiv(B_GRAPHS * HID, TB), TB, 0, stream>>>(emb, 0.0f, B_GRAPHS * HID);
  pool<<<cdiv((long long)N_NODES * (HID / 4), TB), TB, 0, stream>>>(h, attn, batch, emb);
}
